// GPT_15745350107858
// MI455X (gfx1250) — compile-verified, matrix operand order checked by reading, not run on hardware
//
#include <hip/hip_runtime.h>
#include <hip/hip_bf16.h>

// ---------------------------------------------------------------------------
// Types for CDNA5 WMMA (wave32): bf16 A/B fragments (16 bf16 = 8 VGPRs/lane),
// f32 C/D accumulator (8 f32 = 8 VGPRs/lane).
// ---------------------------------------------------------------------------
typedef __attribute__((ext_vector_type(16))) __bf16 bf16x16;
typedef __attribute__((ext_vector_type(8)))  __bf16 bf16x8;
typedef __attribute__((ext_vector_type(8)))  float  floatx8;
typedef __attribute__((ext_vector_type(4)))  unsigned int u32x4;
typedef __attribute__((ext_vector_type(8)))  int i32x8;
typedef __attribute__((ext_vector_type(4)))  int i32x4;

#if defined(__has_builtin)
#if __has_builtin(__builtin_amdgcn_tensor_load_to_lds)
#define USE_TDM 1
#endif
#endif

// Fragment loader for the documented 16-bit 16x32 A layout (and symmetric
// Bt [N,K] layout): lane l holds row (l%16); its 16 elements are two
// contiguous 8-element K runs at K = (l/16)*8 and K = 16 + (l/16)*8.
// Each fragment = two 16-byte loads (global_load_b128 / ds_load_b128).
__device__ __forceinline__ bf16x16 load_frag(const __bf16* base, int stride, int lane) {
    const __bf16* p = base + (size_t)(lane & 15) * stride + ((lane >> 4) << 3);
    bf16x8 lo = *(const bf16x8*)(p);
    bf16x8 hi = *(const bf16x8*)(p + 16);
    bf16x16 r;
#pragma unroll
    for (int i = 0; i < 8; ++i) { r[i] = lo[i]; r[i + 8] = hi[i]; }
    return r;
}

__device__ __forceinline__ floatx8 wmma_bf16(bf16x16 a, bf16x16 b, floatx8 c) {
    return __builtin_amdgcn_wmma_f32_16x16x32_bf16(false, a, false, b, (short)0, c,
                                                   false, false);
}

#ifdef USE_TDM
// ---------------------------------------------------------------------------
// Tensor Data Mover: DMA a 2D tile (tile_rows x 32 bf16, row stride K) from a
// bf16 [rows_total, K] matrix into LDS at lds_off.  D# per ISA 08 §8:
//   group0: count=1 | lds_addr | global_addr(57b) | type=2
//   group1: data_size=1(2B), tensor_dim0=K, tensor_dim1=rows_total,
//           tile_dim0=32, tile_dim1=tile_rows, tensor_dim0_stride=K
// Groups 2/3 unused (2D tile) -> zero.  Tracked by TENSORcnt (one increment
// per instruction, in-order completion).
// ---------------------------------------------------------------------------
__device__ __forceinline__ void tdm_load_2d(const __bf16* gsrc, unsigned lds_off,
                                            int K, int rows_total, int tile_rows) {
    unsigned long long ga = (unsigned long long)(uintptr_t)gsrc;
    u32x4 g0;
    g0[0] = 1u;                                   // count=1, user descriptor
    g0[1] = lds_off;                              // LDS byte address
    g0[2] = (unsigned)ga;                         // global addr [31:0]
    g0[3] = (unsigned)(ga >> 32) | (2u << 30);    // global addr [56:32] | type=2
    i32x8 g1;
    g1[0] = (1 << 16);                            // data_size = 1 (2 bytes)
    g1[1] = (K & 0xFFFF) << 16;                   // tensor_dim0[15:0] @ bits63:48
    g1[2] = (int)(((unsigned)K >> 16) | (((unsigned)rows_total & 0xFFFF) << 16));
    g1[3] = (int)((((unsigned)rows_total) >> 16) | (32u << 16));  // tile_dim0=32
    g1[4] = tile_rows;                            // tile_dim1 (tile_dim2=0)
    g1[5] = K;                                    // tensor_dim0_stride[31:0]
    g1[6] = 0;                                    // stride0 hi / stride1 lo
    g1[7] = 0;
    i32x4 z4 = {};
    i32x8 z8 = {};
    __builtin_amdgcn_tensor_load_to_lds(g0, g1, z4, z4, z8, 0);
}
#endif

// ---------------------------------------------------------------------------
// Weight transpose + f32 -> bf16 convert:  Wt[n*K + k] = (bf16)W[k*N + n]
// ---------------------------------------------------------------------------
__global__ void k_transpose_bf16(const float* __restrict__ W, __bf16* __restrict__ Wt,
                                 int K, int N) {
    __shared__ float tile[32][33];
    int n0 = blockIdx.x * 32, k0 = blockIdx.y * 32;
    int tx = threadIdx.x, ty = threadIdx.y;  // 32 x 8
#pragma unroll
    for (int i = ty; i < 32; i += 8)
        tile[i][tx] = W[(size_t)(k0 + i) * N + n0 + tx];
    __syncthreads();
#pragma unroll
    for (int i = ty; i < 32; i += 8)
        Wt[(size_t)(n0 + i) * K + k0 + tx] = (__bf16)tile[tx][i];
}

// ---------------------------------------------------------------------------
// Embedding: x[b,t,:] = tok_emb[idx[b,t],:] + pos_emb[0,t,:]
// ---------------------------------------------------------------------------
__global__ void k_embed(const int* __restrict__ idx, const float* __restrict__ tok,
                        const float* __restrict__ pos, float* __restrict__ x,
                        int T, int C) {
    int row = blockIdx.x;
    int t = row % T;
    int id = idx[row];
    for (int c = threadIdx.x; c < C; c += blockDim.x)
        x[(size_t)row * C + c] = tok[(size_t)id * C + c] + pos[(size_t)t * C + c];
}

// ---------------------------------------------------------------------------
// LayerNorm (f32 in) -> bf16 out.  One 256-thread block per row.
// ---------------------------------------------------------------------------
__global__ void k_layernorm(const float* __restrict__ x, const float* __restrict__ w,
                            const float* __restrict__ b, __bf16* __restrict__ out, int C) {
    __shared__ float red[2][8];
    int row = blockIdx.x;
    const float* xr = x + (size_t)row * C;
    float s = 0.f, s2 = 0.f;
    for (int c = threadIdx.x; c < C; c += blockDim.x) {
        float v = xr[c];
        s += v; s2 += v * v;
    }
#pragma unroll
    for (int m = 1; m < 32; m <<= 1) {
        s  += __shfl_xor(s,  m, 32);
        s2 += __shfl_xor(s2, m, 32);
    }
    int wid = threadIdx.x >> 5;
    if ((threadIdx.x & 31) == 0) { red[0][wid] = s; red[1][wid] = s2; }
    __syncthreads();
    if (wid == 0) {
        s  = red[0][threadIdx.x & 7];
        s2 = red[1][threadIdx.x & 7];
#pragma unroll
        for (int m = 1; m < 8; m <<= 1) {
            s  += __shfl_xor(s,  m, 32);
            s2 += __shfl_xor(s2, m, 32);
        }
        if (threadIdx.x == 0) { red[0][0] = s; red[1][0] = s2; }
    }
    __syncthreads();
    s = red[0][0]; s2 = red[1][0];
    float mu  = s / C;
    float var = s2 / C - mu * mu;
    float inv = rsqrtf(var + 1e-5f);
    for (int c = threadIdx.x; c < C; c += blockDim.x)
        out[(size_t)row * C + c] = (__bf16)((xr[c] - mu) * inv * w[c] + b[c]);
}

// ---------------------------------------------------------------------------
// TDM-staged WMMA GEMM:  D[M,N] = A[M,K](bf16) * Bt[N,K]^T(bf16) + epilogue.
// 8-wave (256-thread) blocks; block tile 128x128; wave tile 32x64.
// Per 32-wide K step, the Tensor Data Mover DMAs the 128x32 A and B panels
// into LDS (double-buffered, TENSORcnt-synchronized); all 8 waves read
// fragments from LDS (ds_load_b128) and issue 8 wmma each.
// Epilogue modes:
//   0 QKV  : +bias, scatter bf16 q[B,H,T,hd], k[B,H,T,hd], vT[B,H,hd,T]
//   1 GELU : +bias, exact gelu, bf16 out [M,N]
//   2 ADD  : +bias, residual add into f32 out [M,N]
//   3 OUT  : plain f32 out [M,N] (no bias)
// ---------------------------------------------------------------------------
struct GemmP {
    const __bf16* A;
    const __bf16* Bt;
    int M, N, K;
    const float* bias;
    float*  outf;
    __bf16* outb;
    __bf16* q; __bf16* k; __bf16* vT;
    const float* bq; const float* bk; const float* bv;
    int T, H, hd;
};

template <int MODE>
__global__ __launch_bounds__(256) void k_gemm_wmma(GemmP p) {
    __shared__ __align__(16) __bf16 sA[2][128 * 32];
    __shared__ __align__(16) __bf16 sB[2][128 * 32];

    const int lane = threadIdx.x & 31;
    const int wave = threadIdx.x >> 5;
    const int mw = wave >> 1, nw = wave & 1;          // 4 (M) x 2 (N) wave grid
    const int mB = blockIdx.y * 128;
    const int nB = blockIdx.x * 128;
    const int m0 = mB + mw * 32;
    const int n0 = nB + nw * 64;

    floatx8 acc[2][4];
#pragma unroll
    for (int i = 0; i < 2; ++i)
#pragma unroll
        for (int j = 0; j < 4; ++j) acc[i][j] = floatx8{};

    const int nsteps = p.K / 32;

#ifdef USE_TDM
    if (wave == 0) {
        tdm_load_2d(p.A  + (size_t)mB * p.K, (unsigned)(uintptr_t)&sA[0][0], p.K, p.M, 128);
        tdm_load_2d(p.Bt + (size_t)nB * p.K, (unsigned)(uintptr_t)&sB[0][0], p.K, p.N, 128);
    }
    for (int s = 0; s < nsteps; ++s) {
        const int buf = s & 1;
        if (wave == 0) {
            if (s + 1 < nsteps) {
                const int kk = (s + 1) * 32;
                tdm_load_2d(p.A  + (size_t)mB * p.K + kk,
                            (unsigned)(uintptr_t)&sA[buf ^ 1][0], p.K, p.M, 128);
                tdm_load_2d(p.Bt + (size_t)nB * p.K + kk,
                            (unsigned)(uintptr_t)&sB[buf ^ 1][0], p.K, p.N, 128);
                __builtin_amdgcn_s_wait_tensorcnt(2);  // current pair landed
            } else {
                __builtin_amdgcn_s_wait_tensorcnt(0);
            }
        }
        __syncthreads();
        {
            bf16x16 a0 = load_frag(&sA[buf][(mw * 32 + 0)  * 32], 32, lane);
            bf16x16 a1 = load_frag(&sA[buf][(mw * 32 + 16) * 32], 32, lane);
#pragma unroll
            for (int j = 0; j < 4; ++j) {
                bf16x16 b = load_frag(&sB[buf][(nw * 64 + 16 * j) * 32], 32, lane);
                acc[0][j] = wmma_bf16(a0, b, acc[0][j]);
                acc[1][j] = wmma_bf16(a1, b, acc[1][j]);
            }
        }
        __syncthreads();
    }
#else
    for (int s = 0; s < nsteps; ++s) {
        const int kk = s * 32;
        // Cooperative staging: 128x32 bf16 panel = 512 16-byte chunks each.
        for (int c = threadIdx.x; c < 512; c += 256) {
            int row = c >> 2, seg = c & 3;
            *(bf16x8*)&sA[0][row * 32 + seg * 8] =
                *(const bf16x8*)(p.A + (size_t)(mB + row) * p.K + kk + seg * 8);
            *(bf16x8*)&sB[0][row * 32 + seg * 8] =
                *(const bf16x8*)(p.Bt + (size_t)(nB + row) * p.K + kk + seg * 8);
        }
        __syncthreads();
        {
            bf16x16 a0 = load_frag(&sA[0][(mw * 32 + 0)  * 32], 32, lane);
            bf16x16 a1 = load_frag(&sA[0][(mw * 32 + 16) * 32], 32, lane);
#pragma unroll
            for (int j = 0; j < 4; ++j) {
                bf16x16 b = load_frag(&sB[0][(nw * 64 + 16 * j) * 32], 32, lane);
                acc[0][j] = wmma_bf16(a0, b, acc[0][j]);
                acc[1][j] = wmma_bf16(a1, b, acc[1][j]);
            }
        }
        __syncthreads();
    }
#endif

    const int half = lane >> 4;
    const int Cc = p.H * p.hd;
#pragma unroll
    for (int i = 0; i < 2; ++i) {
#pragma unroll
        for (int j = 0; j < 4; ++j) {
            int n = n0 + j * 16 + (lane & 15);
#pragma unroll
            for (int r = 0; r < 8; ++r) {
                int m = m0 + i * 16 + r + 8 * half;
                float v = acc[i][j][r];
                if (MODE == 0) {
                    int which = n / Cc;
                    int nn = n - which * Cc;
                    int hh = nn / p.hd, d = nn - (nn / p.hd) * p.hd;
                    int bb = m / p.T,  t = m - bb * p.T;
                    const float* bp = (which == 0) ? p.bq : (which == 1) ? p.bk : p.bv;
                    __bf16 val = (__bf16)(v + bp[nn]);
                    size_t bh = (size_t)bb * p.H + hh;
                    if (which == 0)      p.q [(bh * p.T + t) * p.hd + d] = val;
                    else if (which == 1) p.k [(bh * p.T + t) * p.hd + d] = val;
                    else                 p.vT[(bh * p.hd + d) * p.T + t] = val;
                } else if (MODE == 1) {
                    float z = v + p.bias[n];
                    float g = 0.5f * z * (1.0f + erff(z * 0.70710678118654752f));
                    p.outb[(size_t)m * p.N + n] = (__bf16)g;
                } else if (MODE == 2) {
                    p.outf[(size_t)m * p.N + n] += v + p.bias[n];
                } else {
                    p.outf[(size_t)m * p.N + n] = v;
                }
            }
        }
    }
}

// ---------------------------------------------------------------------------
// Flash attention, one wave per (b,h, 32 query rows).
// q,k : [B*H, T, hd] bf16;  vT : [B*H, hd, T] bf16;  x += y (residual).
// ---------------------------------------------------------------------------
__global__ __launch_bounds__(32) void k_attention(const __bf16* __restrict__ q,
                                                  const __bf16* __restrict__ k,
                                                  const __bf16* __restrict__ vT,
                                                  float* __restrict__ x,
                                                  int T, int H, int hd, float scale) {
    __shared__ __align__(16) __bf16 plds[32 * 32];

    const int lane = threadIdx.x & 31;
    const int half = lane >> 4;
    const int bh = blockIdx.x;
    const int q0 = blockIdx.y * 32;
    const int bb = bh / H, hh = bh % H;

    const __bf16* qp = q  + (size_t)bh * T * hd;
    const __bf16* kp = k  + (size_t)bh * T * hd;
    const __bf16* vp = vT + (size_t)bh * hd * T;

    bf16x16 qa[2][2];
#pragma unroll
    for (int i = 0; i < 2; ++i)
#pragma unroll
        for (int t = 0; t < 2; ++t)
            qa[i][t] = load_frag(qp + (size_t)(q0 + 16 * i) * hd + 32 * t, hd, lane);

    floatx8 O[2][4];
#pragma unroll
    for (int i = 0; i < 2; ++i)
#pragma unroll
        for (int d = 0; d < 4; ++d) O[i][d] = floatx8{};
    float mrun[2][8], lrun[2][8];
#pragma unroll
    for (int i = 0; i < 2; ++i)
#pragma unroll
        for (int r = 0; r < 8; ++r) { mrun[i][r] = -3.0e38f; lrun[i][r] = 0.f; }

    for (int s0 = 0; s0 <= q0; s0 += 32) {
        floatx8 S[2][2];
#pragma unroll
        for (int i = 0; i < 2; ++i)
#pragma unroll
            for (int j = 0; j < 2; ++j) S[i][j] = floatx8{};
#pragma unroll
        for (int j = 0; j < 2; ++j) {
            bf16x16 kb0 = load_frag(kp + (size_t)(s0 + 16 * j) * hd + 0,  hd, lane);
            bf16x16 kb1 = load_frag(kp + (size_t)(s0 + 16 * j) * hd + 32, hd, lane);
            S[0][j] = wmma_bf16(qa[0][0], kb0, S[0][j]);
            S[0][j] = wmma_bf16(qa[0][1], kb1, S[0][j]);
            S[1][j] = wmma_bf16(qa[1][0], kb0, S[1][j]);
            S[1][j] = wmma_bf16(qa[1][1], kb1, S[1][j]);
        }
#pragma unroll
        for (int i = 0; i < 2; ++i) {
#pragma unroll
            for (int r = 0; r < 8; ++r) {
                int row = q0 + 16 * i + r + 8 * half;
                int c0 = s0 + (lane & 15), c1 = s0 + 16 + (lane & 15);
                float v0 = S[i][0][r] * scale;
                float v1 = S[i][1][r] * scale;
                if (c0 > row) v0 = -3.0e38f;
                if (c1 > row) v1 = -3.0e38f;
                float mx = fmaxf(v0, v1);
#pragma unroll
                for (int msk = 1; msk < 16; msk <<= 1)
                    mx = fmaxf(mx, __shfl_xor(mx, msk, 32));
                float mnew = fmaxf(mrun[i][r], mx);
                float corr = __expf(mrun[i][r] - mnew);
                mrun[i][r] = mnew;
                float p0 = __expf(v0 - mnew);
                float p1 = __expf(v1 - mnew);
                S[i][0][r] = p0; S[i][1][r] = p1;
                float rs = p0 + p1;
#pragma unroll
                for (int msk = 1; msk < 16; msk <<= 1)
                    rs += __shfl_xor(rs, msk, 32);
                lrun[i][r] = lrun[i][r] * corr + rs;
#pragma unroll
                for (int d = 0; d < 4; ++d) O[i][d][r] *= corr;
            }
        }
#pragma unroll
        for (int i = 0; i < 2; ++i)
#pragma unroll
            for (int j = 0; j < 2; ++j)
#pragma unroll
                for (int r = 0; r < 8; ++r)
                    plds[(16 * i + r + 8 * half) * 32 + 16 * j + (lane & 15)] =
                        (__bf16)S[i][j][r];
        __syncthreads();
        bf16x16 pa0 = load_frag(plds,           32, lane);
        bf16x16 pa1 = load_frag(plds + 16 * 32, 32, lane);
#pragma unroll
        for (int d = 0; d < 4; ++d) {
            bf16x16 vb = load_frag(vp + (size_t)(d * 16) * T + s0, T, lane);
            O[0][d] = wmma_bf16(pa0, vb, O[0][d]);
            O[1][d] = wmma_bf16(pa1, vb, O[1][d]);
        }
        __syncthreads();
    }

    const int C = H * hd;
#pragma unroll
    for (int i = 0; i < 2; ++i)
#pragma unroll
        for (int d = 0; d < 4; ++d) {
            int col = hh * hd + d * 16 + (lane & 15);
#pragma unroll
            for (int r = 0; r < 8; ++r) {
                int t = q0 + 16 * i + r + 8 * half;
                x[((size_t)bb * T + t) * C + col] += O[i][d][r] / lrun[i][r];
            }
        }
}

// ---------------------------------------------------------------------------
// Orchestration
// ---------------------------------------------------------------------------
extern "C" void kernel_launch(void* const* d_in, const int* in_sizes, int n_in,
                              void* d_out, int out_size, void* d_ws, size_t ws_size,
                              hipStream_t stream) {
    (void)in_sizes; (void)n_in; (void)out_size; (void)ws_size;
    const int B = 2, T = 1024, C = 1024, L = 8, H = 16, hd = 64, F = 4096, V = 32000;
    const int M = B * T;

    const int*   idx    = (const int*)  d_in[0];
    const float* tok    = (const float*)d_in[1];
    const float* pos    = (const float*)d_in[2];
    const float* ln1w   = (const float*)d_in[3];
    const float* ln1b   = (const float*)d_in[4];
    const float* wq     = (const float*)d_in[5];
    const float* bq     = (const float*)d_in[6];
    const float* wk     = (const float*)d_in[7];
    const float* bk     = (const float*)d_in[8];
    const float* wv     = (const float*)d_in[9];
    const float* bv     = (const float*)d_in[10];
    const float* ln2w   = (const float*)d_in[11];
    const float* ln2b   = (const float*)d_in[12];
    const float* w1     = (const float*)d_in[13];
    const float* b1     = (const float*)d_in[14];
    const float* w2     = (const float*)d_in[15];
    const float* b2     = (const float*)d_in[16];
    const float* lnfw   = (const float*)d_in[17];
    const float* lnfb   = (const float*)d_in[18];
    const float* head_w = (const float*)d_in[19];

    char* ws = (char*)d_ws;
    auto carve = [&](size_t bytes) {
        char* p = ws;
        ws += (bytes + 255) & ~(size_t)255;
        return p;
    };
    float*  x     = (float*) carve((size_t)M * C * 4);
    __bf16* h     = (__bf16*)carve((size_t)M * C * 2);
    __bf16* qb    = (__bf16*)carve((size_t)M * C * 2);
    __bf16* kb    = (__bf16*)carve((size_t)M * C * 2);
    __bf16* vTb   = (__bf16*)carve((size_t)M * C * 2);
    __bf16* mlp   = (__bf16*)carve((size_t)M * F * 2);
    __bf16* WqkvT = (__bf16*)carve((size_t)3 * C * C * 2);
    __bf16* W1T   = (__bf16*)carve((size_t)F * C * 2);
    __bf16* W2T   = (__bf16*)carve((size_t)C * F * 2);
    __bf16* HeadT = (__bf16*)carve((size_t)V * C * 2);

    dim3 tb(32, 8);

    k_transpose_bf16<<<dim3(V / 32, C / 32), tb, 0, stream>>>(head_w, HeadT, C, V);
    k_embed<<<M, 256, 0, stream>>>(idx, tok, pos, x, T, C);

    for (int l = 0; l < L; ++l) {
        k_layernorm<<<M, 256, 0, stream>>>(x, ln1w + (size_t)l * C, ln1b + (size_t)l * C, h, C);

        k_transpose_bf16<<<dim3(C / 32, C / 32), tb, 0, stream>>>(wq + (size_t)l * C * C, WqkvT, C, C);
        k_transpose_bf16<<<dim3(C / 32, C / 32), tb, 0, stream>>>(wk + (size_t)l * C * C, WqkvT + (size_t)C * C, C, C);
        k_transpose_bf16<<<dim3(C / 32, C / 32), tb, 0, stream>>>(wv + (size_t)l * C * C, WqkvT + (size_t)2 * C * C, C, C);

        GemmP pq{};
        pq.A = h; pq.Bt = WqkvT; pq.M = M; pq.N = 3 * C; pq.K = C;
        pq.q = qb; pq.k = kb; pq.vT = vTb;
        pq.bq = bq + (size_t)l * C; pq.bk = bk + (size_t)l * C; pq.bv = bv + (size_t)l * C;
        pq.T = T; pq.H = H; pq.hd = hd;
        k_gemm_wmma<0><<<dim3(3 * C / 128, M / 128), 256, 0, stream>>>(pq);

        k_attention<<<dim3(B * H, T / 32), 32, 0, stream>>>(qb, kb, vTb, x, T, H, hd, 0.125f);

        k_layernorm<<<M, 256, 0, stream>>>(x, ln2w + (size_t)l * C, ln2b + (size_t)l * C, h, C);

        k_transpose_bf16<<<dim3(F / 32, C / 32), tb, 0, stream>>>(w1 + (size_t)l * C * F, W1T, C, F);
        GemmP p1{};
        p1.A = h; p1.Bt = W1T; p1.M = M; p1.N = F; p1.K = C;
        p1.bias = b1 + (size_t)l * F; p1.outb = mlp;
        p1.T = T; p1.H = H; p1.hd = hd;
        k_gemm_wmma<1><<<dim3(F / 128, M / 128), 256, 0, stream>>>(p1);

        k_transpose_bf16<<<dim3(C / 32, F / 32), tb, 0, stream>>>(w2 + (size_t)l * F * C, W2T, F, C);
        GemmP p2{};
        p2.A = mlp; p2.Bt = W2T; p2.M = M; p2.N = C; p2.K = F;
        p2.bias = b2 + (size_t)l * C; p2.outf = x;
        p2.T = T; p2.H = H; p2.hd = hd;
        k_gemm_wmma<2><<<dim3(C / 128, M / 128), 256, 0, stream>>>(p2);
    }

    k_layernorm<<<M, 256, 0, stream>>>(x, lnfw, lnfb, h, C);
    GemmP ph{};
    ph.A = h; ph.Bt = HeadT; ph.M = M; ph.N = V; ph.K = C;
    ph.outf = (float*)d_out;
    ph.T = T; ph.H = H; ph.hd = hd;
    k_gemm_wmma<3><<<dim3(V / 128, M / 128), 256, 0, stream>>>(ph);
}
